// AttentionLayer_66855460930243
// MI455X (gfx1250) — matrix operand order repaired and checked by reading, not compiled, not run on hardware
//
#include <hip/hip_runtime.h>
#include <hip/hip_bf16.h>
#include <math.h>

// ---------------------------------------------------------------------------
// CDNA5 (gfx1250) wave32 WMMA types
// ---------------------------------------------------------------------------
typedef __attribute__((ext_vector_type(16))) __bf16 v16bf;
typedef __attribute__((ext_vector_type(8)))  float  v8f;

union BF16x16 { v16bf v; unsigned int u[8]; };

__device__ __forceinline__ unsigned int f2bf(float f) {
  union { __bf16 h; unsigned short s; } c;
  c.h = (__bf16)f;                 // RNE f32->bf16 convert
  return (unsigned int)c.s;
}

#define D_DIM 128                  // input_dim == attention_dim == 128
#define XS_STRIDE 136              // bf16 elements per LDS row (pad kills bank conflicts)

// ---------------------------------------------------------------------------
// Kernel 0: pack W1 [128x128] f32 into the exact B-operand layout of
// v_wmma_f32_16x16x32_bf16:
//   idx = (((j*4+kb)*32 + lane)*8 + v), value = pack(W1[K][a], W1[K+1][a])
//   K = kb*32 + (lane>=16 ? 8 : 0) + 2*v + (v>=4 ? 8 : 0),  a = j*16 + (lane&15)
// Each wave then fetches B(j,kb) with two coalesced b128 loads.
// ---------------------------------------------------------------------------
__global__ void pack_w1_kernel(const float* __restrict__ W1,
                               unsigned int* __restrict__ w1p) {
  int idx = blockIdx.x * blockDim.x + threadIdx.x;   // 8192 total
  if (idx >= 8192) return;
  int v  = idx & 7;
  int l  = (idx >> 3) & 31;
  int kb = (idx >> 8) & 3;
  int j  = idx >> 10;
  int K  = kb * 32 + ((l & 16) ? 8 : 0) + 2 * v + ((v & 4) ? 8 : 0);
  int a  = j * 16 + (l & 15);
  unsigned int lo = f2bf(W1[K * D_DIM + a]);
  unsigned int hi = f2bf(W1[(K + 1) * D_DIM + a]);
  w1p[idx] = lo | (hi << 16);
}

// ---------------------------------------------------------------------------
// Kernel 1: scores[n] = W2 . tanh(W1^T x_n + b1) + b2 via bf16 WMMA.
// 256 threads = 8 waves; block stages 128 rows of x into LDS as bf16;
// each wave computes one 16-row tile: 8 col-tiles x 4 K-blocks of
// v_wmma_f32_16x16x32_bf16, then tanh + dot with W2, lane-reduce.
// ---------------------------------------------------------------------------
__global__ __launch_bounds__(256) void scores_kernel(
    const float* __restrict__ x,
    const unsigned int* __restrict__ w1p,
    const float* __restrict__ b1,
    const float* __restrict__ W2,
    const float* __restrict__ b2,
    float* __restrict__ scores,
    int Nn) {
  __shared__ unsigned short xs[128 * XS_STRIDE];   // 34 KB bf16 x-tile

  const int tid      = threadIdx.x;
  const int lane     = tid & 31;
  const int wave     = tid >> 5;
  const int blockRow = blockIdx.x * 128;

  // ---- stage x[blockRow : blockRow+128][0:128] as bf16 (coalesced) ----
  #pragma unroll
  for (int i = 0; i < 16; ++i) {
    int f  = tid + i * 256;          // float4 id within tile, 0..4095
    int r  = f >> 5;                 // row in tile
    int c4 = f & 31;                 // float4 column
    int gr = blockRow + r;
    uint2 pk;
    if (gr < Nn) {
      const float4 vv = reinterpret_cast<const float4*>(x)[(size_t)gr * 32 + c4];
      pk.x = f2bf(vv.x) | (f2bf(vv.y) << 16);
      pk.y = f2bf(vv.z) | (f2bf(vv.w) << 16);
    } else {
      pk.x = 0u; pk.y = 0u;
    }
    *reinterpret_cast<uint2*>(&xs[r * XS_STRIDE + c4 * 4]) = pk;
  }
  __syncthreads();

  const int tileRow = blockRow + wave * 16;
  if (tileRow < Nn) {                 // wave-uniform branch: EXEC stays full
    // ---- A operand: row M = lane&15; lanes>=16 carry K+8 / K+24 halves ----
    BF16x16 a[4];
    const int arow = lane & 15;
    const int koff = (lane >> 4) << 3;       // 0 or 8
    #pragma unroll
    for (int kb = 0; kb < 4; ++kb) {
      const uint4 q0 = *reinterpret_cast<const uint4*>(
          &xs[arow * XS_STRIDE + kb * 32 + koff]);
      const uint4 q1 = *reinterpret_cast<const uint4*>(
          &xs[arow * XS_STRIDE + kb * 32 + koff + 16]);
      a[kb].u[0] = q0.x; a[kb].u[1] = q0.y; a[kb].u[2] = q0.z; a[kb].u[3] = q0.w;
      a[kb].u[4] = q1.x; a[kb].u[5] = q1.y; a[kb].u[6] = q1.z; a[kb].u[7] = q1.w;
    }

    float p[8] = {0.f, 0.f, 0.f, 0.f, 0.f, 0.f, 0.f, 0.f};

    #pragma unroll
    for (int j = 0; j < 8; ++j) {
      v8f c = {};
      #pragma unroll
      for (int kb = 0; kb < 4; ++kb) {
        BF16x16 bm;
        const uint4* bp = reinterpret_cast<const uint4*>(
            &w1p[(((j * 4 + kb) * 32) + lane) * 8]);
        const uint4 q0 = bp[0];
        const uint4 q1 = bp[1];
        bm.u[0] = q0.x; bm.u[1] = q0.y; bm.u[2] = q0.z; bm.u[3] = q0.w;
        bm.u[4] = q1.x; bm.u[5] = q1.y; bm.u[6] = q1.z; bm.u[7] = q1.w;
        c = __builtin_amdgcn_wmma_f32_16x16x32_bf16(
            /*neg_a=*/false, a[kb].v, /*neg_b=*/false, bm.v,
            /*c_mod=*/(short)0, c, /*reuse_a=*/false, /*reuse_b=*/false);
      }
      // C layout: VGPR v -> row v (lanes 0-15) / row v+8 (lanes 16-31);
      // col = j*16 + (lane&15)
      const int   col = j * 16 + (lane & 15);
      const float b1v = b1[col];
      const float w2v = W2[col];
      #pragma unroll
      for (int v = 0; v < 8; ++v)
        p[v] += tanhf(c[v] + b1v) * w2v;
    }

    // reduce across the 16 lanes sharing a row (masks<16 stay in each half)
    #pragma unroll
    for (int v = 0; v < 8; ++v) {
      float s = p[v];
      s += __shfl_xor(s, 1, 32);
      s += __shfl_xor(s, 2, 32);
      s += __shfl_xor(s, 4, 32);
      s += __shfl_xor(s, 8, 32);
      p[v] = s;
    }

    const float b2v = b2[0];
    if ((lane & 15) == 0) {
      const int rbase = tileRow + ((lane >> 4) << 3);   // lane0: rows v, lane16: rows v+8
      #pragma unroll
      for (int v = 0; v < 8; ++v) {
        int r = rbase + v;
        if (r < Nn) scores[r] = p[v] + b2v;
      }
    }
  }
}

// ---------------------------------------------------------------------------
// Kernel 2: per-segment softmax + weighted feature pooling.
// batch_index is sorted -> binary-search segment bounds (no atomics).
// 512 threads = 4 row-groups x 128 feature lanes.
// ---------------------------------------------------------------------------
__global__ __launch_bounds__(512) void pool_kernel(
    const float* __restrict__ x,
    const int*   __restrict__ bidx,
    const float* __restrict__ scores,
    float* __restrict__ out,
    int Nn) {
  __shared__ float red[512];
  const int tid = threadIdx.x;
  const int b   = blockIdx.x;

  // lower_bound(b) and lower_bound(b+1)
  int lo = 0, hi = Nn;
  while (lo < hi) { int m = (lo + hi) >> 1; if (bidx[m] < b) lo = m + 1; else hi = m; }
  int lo2 = lo, hi2 = Nn;
  while (lo2 < hi2) { int m = (lo2 + hi2) >> 1; if (bidx[m] < b + 1) lo2 = m + 1; else hi2 = m; }
  const int s0 = lo, s1 = lo2;

  // segment max
  float mx = -3.402823466e+38f;
  for (int i = s0 + tid; i < s1; i += 512) mx = fmaxf(mx, scores[i]);
  red[tid] = mx;
  __syncthreads();
  for (int off = 256; off > 0; off >>= 1) {
    if (tid < off) red[tid] = fmaxf(red[tid], red[tid + off]);
    __syncthreads();
  }
  mx = red[0];
  __syncthreads();

  // denom = sum exp(s - max)
  float sm = 0.f;
  for (int i = s0 + tid; i < s1; i += 512) sm += expf(scores[i] - mx);
  red[tid] = sm;
  __syncthreads();
  for (int off = 256; off > 0; off >>= 1) {
    if (tid < off) red[tid] += red[tid + off];
    __syncthreads();
  }
  const float denom = red[0];
  __syncthreads();

  // out[b][d] = sum_i exp(s_i - max) * x[i][d] / denom
  const int d  = tid & 127;
  const int rg = tid >> 7;          // 0..3
  float acc = 0.f;
  for (int i = s0 + rg; i < s1; i += 4) {
    const float w = expf(scores[i] - mx);
    acc += w * x[(size_t)i * D_DIM + d];
  }
  red[tid] = acc;
  __syncthreads();
  if (rg == 0) {
    float t = red[d] + red[d + 128] + red[d + 256] + red[d + 384];
    out[b * D_DIM + d] = (denom > 0.f) ? (t / denom) : 0.f;
  }
}

// ---------------------------------------------------------------------------
// Launch
// ---------------------------------------------------------------------------
extern "C" void kernel_launch(void* const* d_in, const int* in_sizes, int n_in,
                              void* d_out, int out_size, void* d_ws, size_t ws_size,
                              hipStream_t stream) {
  const float* x    = (const float*)d_in[0];
  const int*   bidx = (const int*)d_in[1];
  const float* W1   = (const float*)d_in[2];
  const float* b1   = (const float*)d_in[3];
  const float* W2   = (const float*)d_in[4];
  const float* b2   = (const float*)d_in[5];
  float*       out  = (float*)d_out;

  const int Nn = in_sizes[0] / D_DIM;       // 500000
  const int B  = out_size / D_DIM;          // 256 segments

  // workspace: [scores: Nn f32][packed W1: 8192 u32]
  float* scores = (float*)d_ws;
  size_t off = ((size_t)Nn * sizeof(float) + 255) & ~(size_t)255;
  unsigned int* w1p = (unsigned int*)((char*)d_ws + off);

  pack_w1_kernel<<<32, 256, 0, stream>>>(W1, w1p);

  const int nTiles  = (Nn + 15) / 16;
  const int nBlocks = (nTiles + 7) / 8;     // 8 waves (16-row tiles) per block
  scores_kernel<<<nBlocks, 256, 0, stream>>>(x, w1p, b1, W2, b2, scores, Nn);

  pool_kernel<<<B, 512, 0, stream>>>(x, bidx, scores, out, Nn);

  (void)n_in; (void)ws_size;
}
